// MaskedCausalAttention_33938831573028
// MI455X (gfx1250) — compile-verified, matrix operand order checked by reading, not run on hardware
//
#include <hip/hip_runtime.h>
#include <hip/hip_bf16.h>

// ---------------------------------------------------------------------------
// MI455X (gfx1250). All matmuls on v_wmma_f32_16x16x32_f16 (f16 in, fp32 acc).
// x and all weights are pre-converted to f16 once; every fragment load is two
// 16-byte (b128) loads matching the ISA 7.12.2 per-lane layout.
// GEMM: each wave produces a 16x64 strip (A-frag reused across 4 WMMAs).
// ---------------------------------------------------------------------------

typedef __attribute__((ext_vector_type(16))) _Float16 v16h;
typedef __attribute__((ext_vector_type(8)))  _Float16 v8h;
typedef __attribute__((ext_vector_type(4)))  _Float16 v4h;
typedef __attribute__((ext_vector_type(8)))  float    v8f;
typedef __attribute__((ext_vector_type(4)))  float    v4f;

#define WMMA_F16(a, b, c) \
  __builtin_amdgcn_wmma_f32_16x16x32_f16(false, (a), false, (b), (short)0, (c), false, false)

#define BB     16
#define TT     768
#define HD     1024
#define ATT_D  768
#define CONV_D 256
#define NH     12
#define DH     64
#define MTOT   (BB * TT)        // 12288
#define WIN    6

// ---------------------------------------------------------------------------
// Fragment loader, contiguous-K (ISA 7.12.2 layout, wave32):
// lane l -> row (l & 15); khi = (l>>4)*8; f[0..7] = K khi..khi+7,
// f[8..15] = K khi+16..khi+23.  Two b128 loads per fragment.
// All call sites guarantee 16-byte alignment (rs and base multiples of 8 halves).
// ---------------------------------------------------------------------------
__device__ __forceinline__ v16h frag_ld(const _Float16* __restrict__ p, int rs) {
  const int lane = threadIdx.x & 31;
  const _Float16* q = p + (lane & 15) * rs + ((lane >> 4) << 3);
  const v8h lo = *(const v8h*)(q);
  const v8h hi = *(const v8h*)(q + 16);
  v16h f;
#pragma unroll
  for (int i = 0; i < 8; ++i) { f[i] = lo[i]; f[8 + i] = hi[i]; }
  return f;
}

// ---------------------------------------------------------------------------
// fp32 -> f16 bulk convert (vectorized, n divisible by 1024)
// ---------------------------------------------------------------------------
__global__ __launch_bounds__(256) void cvt_f32_f16_kernel(
    const float* __restrict__ s, _Float16* __restrict__ d, int n4) {
  const int i = blockIdx.x * 256 + threadIdx.x;
  if (i < n4) {
    const v4f v = ((const v4f*)s)[i];
    v4h h;
#pragma unroll
    for (int j = 0; j < 4; ++j) h[j] = (_Float16)v[j];
    ((v4h*)d)[i] = h;
  }
}

// ---------------------------------------------------------------------------
// GEMM: C[M,N] = A[M,K] @ W[N,K]^T + bias.
// One wave -> 16x64 output strip (4 accumulators, A-frag reused 4x per K-step).
// 8 waves / block cover a 128-row stripe; grid.y over N/64.
// OUT: 0 = fp32 store, 1 = f16 store, 2 = f16 store transposed per head into
// vt[b][h][d][t] (used for the V projection so attention P.V reads contiguously).
// ---------------------------------------------------------------------------
template<bool RELU, int OUT>
__global__ __launch_bounds__(256) void gemm_wt_kernel(
    const _Float16* __restrict__ A, int lda,
    const _Float16* __restrict__ W,        // [N,K] f16 row-major
    const float* __restrict__ bias,        // [N]
    void* __restrict__ Cp, int ldc, int ccol0,
    int K)
{
  const int wid  = threadIdx.x >> 5;
  const int lane = threadIdx.x & 31;
  const int row0 = (blockIdx.x * 8 + wid) * 16;
  const int n0   = blockIdx.y * 64;

  const _Float16* ap = A + (size_t)row0 * lda;
  const _Float16* wp = W + (size_t)n0 * K;

  v8f acc[4] = {};
#pragma unroll 2
  for (int kb = 0; kb < K; kb += 32) {
    const v16h a = frag_ld(ap + kb, lda);
#pragma unroll
    for (int j = 0; j < 4; ++j) {
      const v16h b = frag_ld(wp + (size_t)(j * 16) * K + kb, K);
      acc[j] = WMMA_F16(a, b, acc[j]);
    }
  }

  const int col  = lane & 15;
  const int rofs = (lane >> 4) << 3;        // C layout: lanes 16-31 -> rows +8
#pragma unroll
  for (int j = 0; j < 4; ++j) {
    const int n = n0 + j * 16 + col;
    const float bv = bias[n];
#pragma unroll
    for (int r = 0; r < 8; ++r) {
      float val = acc[j][r] + bv;
      if (RELU) val = fmaxf(val, 0.0f);
      const int m = row0 + rofs + r;
      if (OUT == 2) {
        // vt[b][h][d][t]
        const int b_ = m / TT, t_ = m % TT;
        const int h_ = n / DH, d_ = n % DH;
        ((_Float16*)Cp)[(((size_t)b_ * NH + h_) * DH + d_) * TT + t_] = (_Float16)val;
      } else {
        const size_t off = (size_t)m * ldc + ccol0 + n;
        if (OUT == 1) ((_Float16*)Cp)[off] = (_Float16)val;
        else          ((float*)Cp)[off]    = val;
      }
    }
  }
}

// ---------------------------------------------------------------------------
// Depthwise causal conv branch -> f16 into att2[:, 768:1024].
// ---------------------------------------------------------------------------
__global__ __launch_bounds__(256) void conv_kernel(
    const float* __restrict__ x,
    const float* __restrict__ Krtg, const float* __restrict__ brtg,
    const float* __restrict__ Kobs, const float* __restrict__ bobs,
    const float* __restrict__ Kact, const float* __restrict__ bact,
    _Float16* __restrict__ att2)
{
  const int idx = blockIdx.x * 256 + threadIdx.x;   // B*T*CONV
  const int c  = idx & (CONV_D - 1);
  const int bt = idx >> 8;
  const int t  = bt % TT;
  const int b  = bt / TT;

  const int sel = t % 3;
  const float* Kp = (sel == 0) ? Krtg : (sel == 1) ? Kobs : Kact;
  const float* bp = (sel == 0) ? brtg : (sel == 1) ? bobs : bact;

  float s = bp[c];
#pragma unroll
  for (int w = 0; w < WIN; ++w) {
    const int tt = t + w - (WIN - 1);
    if (tt >= 0)
      s += x[((size_t)(b * TT + tt)) * HD + ATT_D + c] * Kp[c * WIN + w];
  }
  att2[((size_t)(b * TT + t)) * HD + ATT_D + c] = (_Float16)s;
}

// ---------------------------------------------------------------------------
// Flash attention: one wave per (b, head, 16-row q-tile), j-step 32.
// q,k: [B*T, 768] f16;  v: transposed vt[b][h][d][t] f16.
// Online softmax in C layout (16-lane half-wave shfl reductions);
// P staged through LDS to A-frag layout (ds b128 loads).
// ---------------------------------------------------------------------------
__global__ __launch_bounds__(256) void attn_kernel(
    const _Float16* __restrict__ q, const _Float16* __restrict__ k,
    const _Float16* __restrict__ vt, _Float16* __restrict__ att2)
{
  __shared__ _Float16 plds[8][16 * 32];
  const int wid  = threadIdx.x >> 5;
  const int lane = threadIdx.x & 31;
  const int gw    = blockIdx.x * 8 + wid;     // 9216 waves total
  const int itile = gw % (TT / 16);
  const int bh    = gw / (TT / 16);
  const int h     = bh % NH;
  const int b     = bh / NH;
  const int i0    = itile * 16;

  const _Float16* qb  = q  + (size_t)b * TT * ATT_D + h * DH;
  const _Float16* kb  = k  + (size_t)b * TT * ATT_D + h * DH;
  const _Float16* vtb = vt + ((size_t)b * NH + h) * DH * TT;

  v8f acc0 = {}, acc1 = {}, acc2 = {}, acc3 = {};
  float rmax[8], rsum[8];
#pragma unroll
  for (int r = 0; r < 8; ++r) { rmax[r] = -1e30f; rsum[r] = 0.0f; }

  const int col  = lane & 15;
  const int rofs = (lane >> 4) << 3;
  const float scale = 0.125f;                 // 1/sqrt(64)

  for (int j0 = 0; j0 <= i0 + 15; j0 += 32) {
    // ---- S = q . k^T over d=64 (two K=32 steps), two N-tiles ----
    v8f s0 = {}, s1 = {};
#pragma unroll
    for (int kk = 0; kk < DH; kk += 32) {
      const v16h aq = frag_ld(qb + (size_t)i0 * ATT_D + kk, ATT_D);
      const v16h b0 = frag_ld(kb + (size_t)j0 * ATT_D + kk, ATT_D);
      const v16h b1 = frag_ld(kb + (size_t)(j0 + 16) * ATT_D + kk, ATT_D);
      s0 = WMMA_F16(aq, b0, s0);
      s1 = WMMA_F16(aq, b1, s1);
    }
    // ---- mask + online softmax (per-row across 16-lane half-wave) ----
#pragma unroll
    for (int r = 0; r < 8; ++r) {
      const int grow = i0 + rofs + r;
      const float a0 = (j0 + col      <= grow) ? s0[r] * scale : -1e30f;
      const float a1 = (j0 + 16 + col <= grow) ? s1[r] * scale : -1e30f;
      float m = fmaxf(a0, a1);
#pragma unroll
      for (int d = 1; d < 16; d <<= 1) m = fmaxf(m, __shfl_xor(m, d, 32));
      const float nmax = fmaxf(rmax[r], m);
      const float sc = __expf(rmax[r] - nmax);
      rmax[r] = nmax;
      const float p0 = __expf(a0 - nmax);
      const float p1 = __expf(a1 - nmax);
      float ps = p0 + p1;
#pragma unroll
      for (int d = 1; d < 16; d <<= 1) ps += __shfl_xor(ps, d, 32);
      rsum[r] = rsum[r] * sc + ps;
      acc0[r] *= sc; acc1[r] *= sc; acc2[r] *= sc; acc3[r] *= sc;
      plds[wid][(rofs + r) * 32 + col]      = (_Float16)p0;
      plds[wid][(rofs + r) * 32 + 16 + col] = (_Float16)p1;
    }
    // ---- att += P (16x32) . V (32x64), 4 N-tiles; V contiguous via vt ----
    const v16h ap = frag_ld(&plds[wid][0], 32);
    {
      const v16h bv = frag_ld(vtb + (size_t)0  * TT + j0, TT);
      acc0 = WMMA_F16(ap, bv, acc0);
    }
    {
      const v16h bv = frag_ld(vtb + (size_t)16 * TT + j0, TT);
      acc1 = WMMA_F16(ap, bv, acc1);
    }
    {
      const v16h bv = frag_ld(vtb + (size_t)32 * TT + j0, TT);
      acc2 = WMMA_F16(ap, bv, acc2);
    }
    {
      const v16h bv = frag_ld(vtb + (size_t)48 * TT + j0, TT);
      acc3 = WMMA_F16(ap, bv, acc3);
    }
  }

#pragma unroll
  for (int r = 0; r < 8; ++r) {
    const float is = 1.0f / rsum[r];
    const size_t off = ((size_t)(b * TT + i0 + rofs + r)) * HD + h * DH + col;
    att2[off +  0] = (_Float16)(acc0[r] * is);
    att2[off + 16] = (_Float16)(acc1[r] * is);
    att2[off + 32] = (_Float16)(acc2[r] * is);
    att2[off + 48] = (_Float16)(acc3[r] * is);
  }
}

// ---------------------------------------------------------------------------
// gated = softmax(z, axis=-1) * att2   (row width 1024, one wave per row)
// ---------------------------------------------------------------------------
__global__ __launch_bounds__(256) void gate_kernel(
    const float* __restrict__ z, const _Float16* __restrict__ att2,
    _Float16* __restrict__ gated)
{
  const int wid  = threadIdx.x >> 5;
  const int lane = threadIdx.x & 31;
  const size_t m = (size_t)blockIdx.x * 8 + wid;
  const float* zr = z + m * HD;

  float vals[32];
  float mx = -1e30f;
#pragma unroll
  for (int i = 0; i < 32; ++i) { vals[i] = zr[i * 32 + lane]; mx = fmaxf(mx, vals[i]); }
#pragma unroll
  for (int d = 1; d < 32; d <<= 1) mx = fmaxf(mx, __shfl_xor(mx, d, 32));
  float s = 0.0f;
#pragma unroll
  for (int i = 0; i < 32; ++i) { vals[i] = __expf(vals[i] - mx); s += vals[i]; }
#pragma unroll
  for (int d = 1; d < 32; d <<= 1) s += __shfl_xor(s, d, 32);
  const float is = 1.0f / s;
#pragma unroll
  for (int i = 0; i < 32; ++i) {
    const size_t off = m * HD + i * 32 + lane;
    gated[off] = (_Float16)(vals[i] * is * (float)att2[off]);
  }
}

// ---------------------------------------------------------------------------
// Launch
// ---------------------------------------------------------------------------
static inline void cvt(const float* s, _Float16* d, size_t n, hipStream_t st) {
  const int n4 = (int)(n / 4);
  cvt_f32_f16_kernel<<<dim3((n4 + 255) / 256), dim3(256), 0, st>>>(s, d, n4);
}

extern "C" void kernel_launch(void* const* d_in, const int* in_sizes, int n_in,
                              void* d_out, int out_size, void* d_ws, size_t ws_size,
                              hipStream_t stream) {
  (void)in_sizes; (void)n_in; (void)out_size; (void)ws_size;
  const float* x    = (const float*)d_in[0];
  const float* Wq   = (const float*)d_in[1];
  const float* bq   = (const float*)d_in[2];
  const float* Wk   = (const float*)d_in[3];
  const float* bk   = (const float*)d_in[4];
  const float* Wv   = (const float*)d_in[5];
  const float* bv   = (const float*)d_in[6];
  const float* Wf   = (const float*)d_in[7];
  const float* bf   = (const float*)d_in[8];
  const float* Wf1  = (const float*)d_in[9];
  const float* bf1  = (const float*)d_in[10];
  const float* Wf2  = (const float*)d_in[11];
  const float* bf2  = (const float*)d_in[12];
  const float* Wp   = (const float*)d_in[13];
  const float* bp   = (const float*)d_in[14];
  const float* Krtg = (const float*)d_in[15];
  const float* brtg = (const float*)d_in[16];
  const float* Kobs = (const float*)d_in[17];
  const float* bobs = (const float*)d_in[18];
  const float* Kact = (const float*)d_in[19];
  const float* bact = (const float*)d_in[20];

  // Workspace layout (~205 MB)
  _Float16* xh    = (_Float16*)d_ws;                       // [12288, 1024]
  _Float16* qf    = xh   + (size_t)MTOT * HD;              // [12288, 768]
  _Float16* kf    = qf   + (size_t)MTOT * ATT_D;           // [12288, 768]
  _Float16* vt    = kf   + (size_t)MTOT * ATT_D;           // [B, NH, DH, TT]
  _Float16* att2  = vt   + (size_t)MTOT * ATT_D;           // [12288, 1024]
  _Float16* xbar  = att2 + (size_t)MTOT * HD;              // [12288, 512]
  float*    zbuf  = (float*)(xbar + (size_t)MTOT * 512);   // [12288, 1024] fp32
  _Float16* gated = (_Float16*)(zbuf + (size_t)MTOT * HD); // [12288, 1024]
  _Float16* Whq   = gated + (size_t)MTOT * HD;
  _Float16* Whk   = Whq + (size_t)ATT_D * ATT_D;
  _Float16* Whv   = Whk + (size_t)ATT_D * ATT_D;
  _Float16* Whf   = Whv + (size_t)ATT_D * ATT_D;           // [512, 1024]
  _Float16* Whf1  = Whf + (size_t)512 * HD;                // [768, 512]
  _Float16* Whf2  = Whf1 + (size_t)ATT_D * 512;            // [256, 512]
  _Float16* Whp   = Whf2 + (size_t)CONV_D * 512;           // [1024, 1024]

  // ---- one-time f16 staging of activations and weights ----
  cvt(x,   xh,   (size_t)MTOT * HD,     stream);
  cvt(Wq,  Whq,  (size_t)ATT_D * ATT_D, stream);
  cvt(Wk,  Whk,  (size_t)ATT_D * ATT_D, stream);
  cvt(Wv,  Whv,  (size_t)ATT_D * ATT_D, stream);
  cvt(Wf,  Whf,  (size_t)512 * HD,      stream);
  cvt(Wf1, Whf1, (size_t)ATT_D * 512,   stream);
  cvt(Wf2, Whf2, (size_t)CONV_D * 512,  stream);
  cvt(Wp,  Whp,  (size_t)HD * HD,       stream);

  const dim3 blk(256);
  const int mb = MTOT / 128;   // 96 blocks of 8 waves x 16 rows

  // q, k = x1 @ W^T + b (f16 out); v stored transposed per head
  gemm_wt_kernel<false, 1><<<dim3(mb, ATT_D / 64), blk, 0, stream>>>(
      xh, HD, Whq, bq, qf, ATT_D, 0, ATT_D);
  gemm_wt_kernel<false, 1><<<dim3(mb, ATT_D / 64), blk, 0, stream>>>(
      xh, HD, Whk, bk, kf, ATT_D, 0, ATT_D);
  gemm_wt_kernel<false, 2><<<dim3(mb, ATT_D / 64), blk, 0, stream>>>(
      xh, HD, Whv, bv, vt, 0, 0, ATT_D);

  // conv branch -> att2[:, 768:1024]
  conv_kernel<<<dim3((BB * TT * CONV_D) / 256), blk, 0, stream>>>(
      x, Krtg, brtg, Kobs, bobs, Kact, bact, att2);

  // attention -> att2[:, 0:768]
  attn_kernel<<<dim3((BB * NH * (TT / 16)) / 8), blk, 0, stream>>>(qf, kf, vt, att2);

  // xbar = relu(att2 @ Wf^T + bf)
  gemm_wt_kernel<true, 1><<<dim3(mb, 512 / 64), blk, 0, stream>>>(
      att2, HD, Whf, bf, xbar, 512, 0, HD);

  // z = [xbar @ Wf1^T + bf1 | xbar @ Wf2^T + bf2]  (fp32)
  gemm_wt_kernel<false, 0><<<dim3(mb, ATT_D / 64), blk, 0, stream>>>(
      xbar, 512, Whf1, bf1, zbuf, HD, 0, 512);
  gemm_wt_kernel<false, 0><<<dim3(mb, CONV_D / 64), blk, 0, stream>>>(
      xbar, 512, Whf2, bf2, zbuf, HD, ATT_D, 512);

  // gated = softmax(z) * att2
  gate_kernel<<<dim3(MTOT / 8), blk, 0, stream>>>(zbuf, att2, gated);

  // out = gated @ Wp^T + bp  (fp32 out)
  gemm_wt_kernel<false, 0><<<dim3(mb, HD / 64), blk, 0, stream>>>(
      gated, HD, Whp, bp, (float*)d_out, HD, 0, HD);
}
